// GAT_22960895164528
// MI455X (gfx1250) — matrix operand order repaired and checked by reading, not compiled, not run on hardware
//
#include <hip/hip_runtime.h>
#include <hip/hip_bf16.h>

#define N_NODES   50000
#define N_EDGES   800000
#define F         128      // IN_C == HEADS*HID
#define HEADS     4
#define HID       32
#define G_GRAPHS  512
#define OUT_CH    100
#define NEG_SLOPE 0.2f
#define EPS_F     1e-16f

typedef __attribute__((ext_vector_type(8)))  float        v8f;
typedef __attribute__((ext_vector_type(16))) __bf16       v16bf;
typedef __attribute__((ext_vector_type(8)))  unsigned int v8u;

// ---------- helpers ----------
__device__ __forceinline__ unsigned int pack_bf2(float lo, float hi) {
  // round-to-nearest-even f32 -> bf16, packed pair
  unsigned int a = __float_as_uint(lo);
  unsigned int b = __float_as_uint(hi);
  a = a + 0x7FFFu + ((a >> 16) & 1u);
  b = b + 0x7FFFu + ((b >> 16) & 1u);
  return (a >> 16) | (b & 0xFFFF0000u);
}

// order-preserving float <-> uint encoding for atomicMax on floats
__device__ __forceinline__ unsigned int fenc(float f) {
  unsigned int u = __float_as_uint(f);
  return (u & 0x80000000u) ? ~u : (u | 0x80000000u);
}
__device__ __forceinline__ float fdec(unsigned int u) {
  return (u & 0x80000000u) ? __uint_as_float(u & 0x7FFFFFFFu)
                           : __uint_as_float(~u);
}

// ---------- K0: bulk f32 -> packed bf16 conversion (one pass, HBM-bound) ----------
__global__ void f32_to_bf16_pairs(const float* __restrict__ in,
                                  unsigned int* __restrict__ out, int npairs) {
  int t = blockIdx.x * blockDim.x + threadIdx.x;
  if (t >= npairs) return;
  float2 f = ((const float2*)in)[t];
  out[t] = pack_bf2(f.x, f.y);
}

// ---------- K1: h = x @ W^T via WMMA (bf16 in, f32 acc) ----------
// Operands pre-converted to bf16 row-major; all fragment loads are contiguous
// 16B dwordx4 loads (A: lanes 0-15 K0-7/16-23, lanes 16-31 K8-15/24-31;
// B: same K order as A). Inner loop = loads + wmma.
__global__ void __launch_bounds__(256)
gat_gemm_wmma(const __bf16* __restrict__ Xb, const __bf16* __restrict__ Wb,
              float* __restrict__ H, int nTiles) {
  const int wave = blockIdx.x * (blockDim.x >> 5) + (threadIdx.x >> 5);
  if (wave >= nTiles) return;                 // wave-uniform: EXEC stays all-ones
  const int lane = threadIdx.x & 31;
  const int mrow = lane & 15;
  const int hi   = lane >> 4;
  const int row0 = wave * 16;

  // A fragments: row = node, 4 k-tiles of 32; per tile two uint4 loads
  v8u au[4];
  const uint4* xrow = (const uint4*)(Xb + (size_t)(row0 + mrow) * F);
#pragma unroll
  for (int kt = 0; kt < 4; ++kt) {
    uint4 lo  = xrow[kt * 4 + hi];       // dwords kt*16 + hi*4 .. +3  (K lo half)
    uint4 hi4 = xrow[kt * 4 + 2 + hi];   // dwords kt*16 + 8 + hi*4   (K hi half)
    au[kt] = (v8u){lo.x, lo.y, lo.z, lo.w, hi4.x, hi4.y, hi4.z, hi4.w};
  }

  for (int jt = 0; jt < 8; ++jt) {            // 8 output-feature tiles of 16
    v8f c = {0.f, 0.f, 0.f, 0.f, 0.f, 0.f, 0.f, 0.f};
    const uint4* wrow = (const uint4*)(Wb + (size_t)(jt * 16 + mrow) * F);
#pragma unroll
    for (int kt = 0; kt < 4; ++kt) {
      uint4 b0 = wrow[kt * 4 + hi];           // dwords kt*16 + hi*4 .. +3  (K lo half)
      uint4 b1 = wrow[kt * 4 + 2 + hi];       // dwords kt*16 + 8 + hi*4   (K hi half)
      v8u bu = (v8u){b0.x, b0.y, b0.z, b0.w, b1.x, b1.y, b1.z, b1.w};
      v16bf av = __builtin_bit_cast(v16bf, au[kt]);
      v16bf bv = __builtin_bit_cast(v16bf, bu);
      c = __builtin_amdgcn_wmma_f32_16x16x32_bf16(
              false, av, false, bv, (short)0, c, false, false);
    }
#pragma unroll
    for (int r = 0; r < 8; ++r)
      H[(size_t)(row0 + hi * 8 + r) * F + jt * 16 + mrow] = c[r];
  }
}

// ---------- K1b: per-(node,head) attention coefficients ----------
__global__ void att_coeff(const float* __restrict__ H,
                          const float* __restrict__ att_s,
                          const float* __restrict__ att_d,
                          float* __restrict__ a_src, float* __restrict__ a_dst) {
  int t = blockIdx.x * blockDim.x + threadIdx.x;
  if (t >= N_NODES * HEADS) return;
  int n = t >> 2, q = t & 3;
  const float* hp = H + (size_t)n * F + q * HID;
  const float* sp = att_s + q * HID;
  const float* dp = att_d + q * HID;
  float s1 = 0.f, s2 = 0.f;
#pragma unroll
  for (int c = 0; c < HID; ++c) { float hv = hp[c]; s1 += hv * sp[c]; s2 += hv * dp[c]; }
  a_src[t] = s1;
  a_dst[t] = s2;
}

// ---------- K2: segment max over incoming edges ----------
__global__ void edge_max(const int* __restrict__ ei,
                         const float* __restrict__ a_src, const float* __restrict__ a_dst,
                         unsigned int* __restrict__ m, int E_, int Etot) {
  int e = blockIdx.x * blockDim.x + threadIdx.x;
  if (e >= Etot) return;
  int s, d;
  if (e < E_) { s = ei[e]; d = ei[E_ + e]; } else { s = e - E_; d = s; }
#pragma unroll
  for (int q = 0; q < HEADS; ++q) {
    float al = a_src[s * HEADS + q] + a_dst[d * HEADS + q];
    al = al > 0.f ? al : NEG_SLOPE * al;
    atomicMax(&m[d * HEADS + q], fenc(al));
  }
}

// ---------- K3: softmax denominator ----------
__global__ void edge_denom(const int* __restrict__ ei,
                           const float* __restrict__ a_src, const float* __restrict__ a_dst,
                           const unsigned int* __restrict__ m,
                           float* __restrict__ denom, int E_, int Etot) {
  int e = blockIdx.x * blockDim.x + threadIdx.x;
  if (e >= Etot) return;
  int s, d;
  if (e < E_) { s = ei[e]; d = ei[E_ + e]; } else { s = e - E_; d = s; }
#pragma unroll
  for (int q = 0; q < HEADS; ++q) {
    float al = a_src[s * HEADS + q] + a_dst[d * HEADS + q];
    al = al > 0.f ? al : NEG_SLOPE * al;
    float p = __expf(al - fdec(m[d * HEADS + q]));
    atomicAdd(&denom[d * HEADS + q], p);
  }
}

// ---------- K4: weighted aggregation, one wave per edge ----------
// Lane c covers feature columns c, c+32, c+64, c+96 (one per head) -> fully
// coalesced 128B gather of h[src] and coalesced atomic adds into AGG[dst].
__global__ void edge_aggregate(const int* __restrict__ ei,
                               const float* __restrict__ a_src, const float* __restrict__ a_dst,
                               const unsigned int* __restrict__ m,
                               const float* __restrict__ denom,
                               const float* __restrict__ H, float* __restrict__ AGG,
                               int E_, int Etot) {
  int gid  = blockIdx.x * blockDim.x + threadIdx.x;
  int e    = gid >> 5;
  int lane = gid & 31;
  if (e >= Etot) return;
  int s, d;
  if (e < E_) { s = ei[e]; d = ei[E_ + e]; } else { s = e - E_; d = s; }
  float w[HEADS];
#pragma unroll
  for (int q = 0; q < HEADS; ++q) {
    float al = a_src[s * HEADS + q] + a_dst[d * HEADS + q];
    al = al > 0.f ? al : NEG_SLOPE * al;
    float p = __expf(al - fdec(m[d * HEADS + q]));
    w[q] = p / (denom[d * HEADS + q] + EPS_F);
  }
  const float* hs = H + (size_t)s * F;
  float* ag = AGG + (size_t)d * F;
#pragma unroll
  for (int q = 0; q < HEADS; ++q) {
    int f = q * HID + lane;
    atomicAdd(ag + f, w[q] * hs[f]);
  }
}

// ---------- K5: bias + ELU; writes f32 in-place (AGG) + packed bf16 ----------
__global__ void finalize_elu(float* __restrict__ AGG, const float* __restrict__ bias,
                             unsigned int* __restrict__ Obf) {
  int t = blockIdx.x * blockDim.x + threadIdx.x;      // one thread per pair
  if (t >= N_NODES * F / 2) return;
  int j = (2 * t) & (F - 1);
  float2 v = ((const float2*)AGG)[t];
  float v0 = v.x + bias[j];
  float v1 = v.y + bias[j + 1];
  v0 = v0 > 0.f ? v0 : (__expf(v0) - 1.f);
  v1 = v1 > 0.f ? v1 : (__expf(v1) - 1.f);
  ((float2*)AGG)[t] = make_float2(v0, v1);
  Obf[t] = pack_bf2(v0, v1);
}

// ---------- K6: global mean pool (sums + counts) ----------
__global__ void pool_kernel(const float* __restrict__ FEAT, const int* __restrict__ batch,
                            float* __restrict__ sums, float* __restrict__ cnt) {
  int t = blockIdx.x * blockDim.x + threadIdx.x;
  if (t >= N_NODES * F) return;
  int n = t >> 7, j = t & 127;
  int g = batch[n];
  atomicAdd(&sums[g * F + j], FEAT[t]);
  if (j == 0) atomicAdd(&cnt[g], 1.0f);
}

// ---------- K7: final FC  out = (sums/cnt) @ fcW^T + fcb ----------
__global__ void fc_kernel(const float* __restrict__ sums, const float* __restrict__ cnt,
                          const float* __restrict__ Wfc, const float* __restrict__ bfc,
                          float* __restrict__ out) {
  int t = blockIdx.x * blockDim.x + threadIdx.x;
  if (t >= G_GRAPHS * OUT_CH) return;
  int g = t / OUT_CH, oc = t - g * OUT_CH;
  float cg = cnt[g];
  cg = cg > 1.f ? cg : 1.f;
  const float* sp = sums + (size_t)g * F;
  const float* wp = Wfc + (size_t)oc * F;
  float acc = 0.f;
#pragma unroll 4
  for (int k = 0; k < F; ++k) acc += sp[k] * wp[k];
  out[t] = acc / cg + bfc[oc];
}

// ---------- launch ----------
extern "C" void kernel_launch(void* const* d_in, const int* in_sizes, int n_in,
                              void* d_out, int out_size, void* d_ws, size_t ws_size,
                              hipStream_t stream) {
  const float* x     = (const float*)d_in[0];
  const int*   ei    = (const int*)d_in[1];
  const int*   batch = (const int*)d_in[2];
  const float* Wl[3]  = {(const float*)d_in[3],  (const float*)d_in[7],  (const float*)d_in[11]};
  const float* asl[3] = {(const float*)d_in[4],  (const float*)d_in[8],  (const float*)d_in[12]};
  const float* adl[3] = {(const float*)d_in[5],  (const float*)d_in[9],  (const float*)d_in[13]};
  const float* bl[3]  = {(const float*)d_in[6],  (const float*)d_in[10], (const float*)d_in[14]};
  const float* fcW = (const float*)d_in[15];
  const float* fcb = (const float*)d_in[16];
  float* out = (float*)d_out;

  char* ws = (char*)d_ws;
  size_t off = 0;
  auto alloc = [&](size_t bytes) -> void* {
    void* p = ws + off;
    off += (bytes + 255) & ~(size_t)255;
    return p;
  };
  __bf16*   Pbf   = (__bf16*)alloc((size_t)N_NODES * F * sizeof(__bf16));
  __bf16*   Qbf   = (__bf16*)alloc((size_t)N_NODES * F * sizeof(__bf16));
  float*    H     = (float*)alloc((size_t)N_NODES * F * sizeof(float));
  float*    AGG   = (float*)alloc((size_t)N_NODES * F * sizeof(float));
  float*    a_src = (float*)alloc((size_t)N_NODES * HEADS * sizeof(float));
  float*    a_dst = (float*)alloc((size_t)N_NODES * HEADS * sizeof(float));
  unsigned* mEnc  = (unsigned*)alloc((size_t)N_NODES * HEADS * sizeof(unsigned));
  float*    den   = (float*)alloc((size_t)N_NODES * HEADS * sizeof(float));
  float*    sums  = (float*)alloc((size_t)G_GRAPHS * F * sizeof(float));
  float*    cnt   = (float*)alloc((size_t)G_GRAPHS * sizeof(float));
  __bf16*   Wbf[3];
  for (int i = 0; i < 3; ++i)
    Wbf[i] = (__bf16*)alloc((size_t)F * F * sizeof(__bf16));
  (void)in_sizes; (void)n_in; (void)out_size; (void)ws_size;

  const int Etot   = N_EDGES + N_NODES;       // self-loops appended
  const int nTiles = N_NODES / 16;            // 3125
  const int xPairs = N_NODES * F / 2;
  const int wPairs = F * F / 2;

  // one-time conversions (bf16 operands for WMMA)
  f32_to_bf16_pairs<<<(xPairs + 255) / 256, 256, 0, stream>>>(x, (unsigned*)Pbf, xPairs);
  for (int i = 0; i < 3; ++i)
    f32_to_bf16_pairs<<<(wPairs + 255) / 256, 256, 0, stream>>>(Wl[i], (unsigned*)Wbf[i], wPairs);

  for (int layer = 0; layer < 3; ++layer) {
    const __bf16* in_bf = (layer == 2) ? Qbf : Pbf;       // L1:Pbf L2:Pbf L3:Qbf
    __bf16*       out_bf = (layer == 1) ? Qbf : Pbf;      // L1:Pbf L2:Qbf L3:Pbf

    gat_gemm_wmma<<<(nTiles + 7) / 8, 256, 0, stream>>>(in_bf, Wbf[layer], H, nTiles);
    att_coeff<<<(N_NODES * HEADS + 255) / 256, 256, 0, stream>>>(
        H, asl[layer], adl[layer], a_src, a_dst);

    hipMemsetAsync(mEnc, 0, (size_t)N_NODES * HEADS * sizeof(unsigned), stream); // 0 < fenc(x) for all x
    hipMemsetAsync(den,  0, (size_t)N_NODES * HEADS * sizeof(float), stream);
    hipMemsetAsync(AGG,  0, (size_t)N_NODES * F * sizeof(float), stream);

    edge_max<<<(Etot + 255) / 256, 256, 0, stream>>>(ei, a_src, a_dst, mEnc, N_EDGES, Etot);
    edge_denom<<<(Etot + 255) / 256, 256, 0, stream>>>(ei, a_src, a_dst, mEnc, den, N_EDGES, Etot);
    {
      long long thr = (long long)Etot * 32;
      edge_aggregate<<<(unsigned)((thr + 255) / 256), 256, 0, stream>>>(
          ei, a_src, a_dst, mEnc, den, H, AGG, N_EDGES, Etot);
    }
    // ELU: f32 result stays in AGG (used for pooling after last layer),
    // bf16 copy feeds the next layer's WMMA GEMM.
    finalize_elu<<<(N_NODES * F / 2 + 255) / 256, 256, 0, stream>>>(
        AGG, bl[layer], (unsigned*)out_bf);
  }

  hipMemsetAsync(sums, 0, (size_t)G_GRAPHS * F * sizeof(float), stream);
  hipMemsetAsync(cnt,  0, (size_t)G_GRAPHS * sizeof(float), stream);
  pool_kernel<<<(N_NODES * F + 255) / 256, 256, 0, stream>>>(AGG, batch, sums, cnt);
  fc_kernel<<<(G_GRAPHS * OUT_CH + 255) / 256, 256, 0, stream>>>(sums, cnt, fcW, fcb, out);
}
